// TritonGatherConv_73254962201305
// MI455X (gfx1250) — matrix operand order repaired and checked by reading, not compiled, not run on hardware
//
#include <hip/hip_runtime.h>
#include <math.h>

// Problem constants (match reference)
#define BB 4
#define LL 1024
#define CC 1024
#define HH 8
#define DD 128
#define KK 64
#define SS 33
#define HALF_S 16
#define MM (BB * LL)      // 4096 rows

typedef __attribute__((ext_vector_type(16))) __bf16 v16bf;
typedef __attribute__((ext_vector_type(8)))  __bf16 v8bf;
typedef __attribute__((ext_vector_type(8)))  float  v8f;

__device__ __forceinline__ unsigned short f2bf(float f) {
    unsigned int u = __float_as_uint(f);
    u += 0x7FFFu + ((u >> 16) & 1u);        // round-to-nearest-even
    return (unsigned short)(u >> 16);
}
__device__ __forceinline__ float silu_f(float v) {
    return v / (1.0f + __expf(-v));
}

// A-fragment load (contiguous k order): per lane two 16B chunks at
// +hi*16 and +hi*16+8 merged into a v16bf.
__device__ __forceinline__ v16bf load_afrag(const unsigned short* p, int k0) {
    v8bf lo = *reinterpret_cast<const v8bf*>(p + k0);
    v8bf hi = *reinterpret_cast<const v8bf*>(p + k0 + 8);
    return __builtin_shufflevector(lo, hi,
        0, 1, 2, 3, 4, 5, 6, 7, 8, 9, 10, 11, 12, 13, 14, 15);
}

// One K-step fragment set: 2 A subtiles (32 M rows) x 4 B subtiles (64 N cols)
struct Frags {
    v16bf a0, a1;
    v16bf b[4];
};

__device__ __forceinline__ void load_frags(Frags& f,
        const unsigned short* aptr0, const unsigned short* aptr1,
        const unsigned short* bp0, const unsigned short* bp1,
        const unsigned short* bp2, const unsigned short* bp3, int k0) {
    f.a0 = load_afrag(aptr0, k0);
    f.a1 = load_afrag(aptr1, k0);
    f.b[0] = *reinterpret_cast<const v16bf*>(bp0 + k0);
    f.b[1] = *reinterpret_cast<const v16bf*>(bp1 + k0);
    f.b[2] = *reinterpret_cast<const v16bf*>(bp2 + k0);
    f.b[3] = *reinterpret_cast<const v16bf*>(bp3 + k0);
}

__device__ __forceinline__ void wmma_step(v8f acc[2][4], const Frags& f) {
    #pragma unroll
    for (int j = 0; j < 4; ++j)
        acc[0][j] = __builtin_amdgcn_wmma_f32_16x16x32_bf16(
                        false, f.a0, false, f.b[j], (short)0, acc[0][j],
                        false, false);
    #pragma unroll
    for (int j = 0; j < 4; ++j)
        acc[1][j] = __builtin_amdgcn_wmma_f32_16x16x32_bf16(
                        false, f.a1, false, f.b[j], (short)0, acc[1][j],
                        false, false);
}

// ---------------------------------------------------------------------------
// fp32 -> bf16 conversion (grid-stride)
// ---------------------------------------------------------------------------
__global__ void cvt_f32_to_bf16(const float* __restrict__ in,
                                unsigned short* __restrict__ out, int n) {
    int i = blockIdx.x * blockDim.x + threadIdx.x;
    int stride = gridDim.x * blockDim.x;
    for (; i < n; i += stride) out[i] = f2bf(in[i]);
}

// ---------------------------------------------------------------------------
// bf16 WMMA GEMM:  out[m,n] = silu( sum_k A[m,k]*W[n,k] + bias[n] )
// A: M x Kd row-major bf16, W: N x Kd row-major bf16 (computes A @ W^T).
// Block = 256 threads = 8 waves: 2 along M x 4 along N.
// Wave tile = 32(M) x 64(N): 2 A-frags, 4 B-frags, 8 WMMAs per K-step.
// Block tile = 64(M) x 256(N).
// Ping-pong double-buffered fragments (2 sets, K unrolled by 64) so loads
// overlap v_wmma issue with no register-rotation moves.
// Requires Kd a multiple of 64 (Kd == 1024 here).
// ---------------------------------------------------------------------------
__global__ __launch_bounds__(256)
void gemm_bf16_silu(const unsigned short* __restrict__ A,
                    const unsigned short* __restrict__ W,
                    const float* __restrict__ bias,
                    float* __restrict__ out,
                    int M, int N, int Kd) {
    const int lane = threadIdx.x & 31;
    const int wave = threadIdx.x >> 5;
    const int lr   = lane & 15;
    const int hi   = lane >> 4;
    const int wm   = wave & 1;   // 2 waves along M
    const int wn   = wave >> 1;  // 4 waves along N
    const int mBase = blockIdx.y * 64 + wm * 32;
    const int nBase = blockIdx.x * 256 + wn * 64;

    // A fragment pointers for the two 16-row subtiles
    const unsigned short* ap0 = A + (size_t)(mBase + lr) * Kd + hi * 16;
    const unsigned short* ap1 = A + (size_t)(mBase + 16 + lr) * Kd + hi * 16;

    // B fragment pointers: row n (clamped for small-N launches), 16
    // contiguous K values per lane at +hi*16 (32B aligned load)
    int n0 = nBase + lr;      if (n0 > N - 1) n0 = N - 1;
    int n1 = nBase + 16 + lr; if (n1 > N - 1) n1 = N - 1;
    int n2 = nBase + 32 + lr; if (n2 > N - 1) n2 = N - 1;
    int n3 = nBase + 48 + lr; if (n3 > N - 1) n3 = N - 1;
    const unsigned short* bp0 = W + (size_t)n0 * Kd + hi * 16;
    const unsigned short* bp1 = W + (size_t)n1 * Kd + hi * 16;
    const unsigned short* bp2 = W + (size_t)n2 * Kd + hi * 16;
    const unsigned short* bp3 = W + (size_t)n3 * Kd + hi * 16;

    v8f acc[2][4];
    #pragma unroll
    for (int t = 0; t < 2; ++t)
        #pragma unroll
        for (int j = 0; j < 4; ++j)
            acc[t][j] = (v8f){0.f, 0.f, 0.f, 0.f, 0.f, 0.f, 0.f, 0.f};

    Frags f0, f1;
    load_frags(f0, ap0, ap1, bp0, bp1, bp2, bp3, 0);

    // Ping-pong over 64-K double-steps; pointers advance, element offsets
    // 0 / +32 fold into the global_load immediate offset.
    int iters = Kd / 64;                 // Kd == 1024 -> 16
    for (int it = 0; it < iters - 1; ++it) {
        __builtin_prefetch(ap0 + 128, 0, 3);
        load_frags(f1, ap0, ap1, bp0, bp1, bp2, bp3, 32);
        wmma_step(acc, f0);
        ap0 += 64; ap1 += 64; bp0 += 64; bp1 += 64; bp2 += 64; bp3 += 64;
        load_frags(f0, ap0, ap1, bp0, bp1, bp2, bp3, 0);
        wmma_step(acc, f1);
    }
    // Peeled last double-step
    load_frags(f1, ap0, ap1, bp0, bp1, bp2, bp3, 32);
    wmma_step(acc, f0);
    wmma_step(acc, f1);

    // Store: D layout VGPR r -> M = r + 8*hi, N = lr; bias + silu.
    #pragma unroll
    for (int t = 0; t < 2; ++t) {
        #pragma unroll
        for (int j = 0; j < 4; ++j) {
            int n = nBase + j * 16 + lr;
            if (n >= N) continue;
            float bv = bias ? bias[n] : 0.0f;
            #pragma unroll
            for (int r = 0; r < 8; ++r) {
                int m = mBase + t * 16 + hi * 8 + r;
                out[(size_t)m * N + n] = silu_f(acc[t][j][r] + bv);
            }
        }
    }
}

// ---------------------------------------------------------------------------
// freq_avg / phase_avg from silu'd wave projection (M rows x 16 cols)
// cols 0..7 -> freq = sigmoid*15+1 ; cols 8..15 -> phase = tanh*16
// ---------------------------------------------------------------------------
__global__ void freq_phase_kernel(const float* __restrict__ wv,
                                  float* __restrict__ freq_avg,
                                  float* __restrict__ phase_avg, int n) {
    int t = blockIdx.x * blockDim.x + threadIdx.x;
    if (t >= n) return;
    const float* p = wv + (size_t)t * 16;
    float fs = 0.f, ps = 0.f;
    #pragma unroll
    for (int h = 0; h < 8; ++h)
        fs += 1.0f / (1.0f + __expf(-p[h])) * 15.0f + 1.0f;
    #pragma unroll
    for (int h = 0; h < 8; ++h)
        ps += tanhf(p[8 + h]) * 16.0f;
    freq_avg[t]  = fs * 0.125f;
    phase_avg[t] = ps * 0.125f;
}

// ---------------------------------------------------------------------------
// Bilinear gather + per-head S-tap conv.
// One block per (b,l,h); 128 threads = d. Reads x (fp32, L2-resident),
// writes out_h as bf16 for the final WMMA GEMM.
// ---------------------------------------------------------------------------
__global__ __launch_bounds__(128)
void gather_conv_kernel(const float* __restrict__ x,      // B,L,C fp32
                        const float* __restrict__ wker,   // (B*L) x 512 silu'd
                        const float* __restrict__ freq_avg,
                        const float* __restrict__ phase_avg,
                        unsigned short* __restrict__ outh) { // (B*L) x C bf16
    int blh = blockIdx.x;           // b*L*H + l*H + h
    int h   = blh % HH;
    int bl  = blh / HH;             // b*L + l
    int l   = bl % LL;
    int b   = bl / LL;
    int d   = threadIdx.x;

    float fa = freq_avg[bl];
    float pa = phase_avg[bl];
    const float* xb = x + (size_t)b * LL * CC + h * DD + d;
    const float* wp = wker + (size_t)bl * (HH * KK) + h * KK;

    float acc = 0.0f;
    for (int s = 0; s < SS; ++s) {
        float pos = (float)l + (float)(s - HALF_S) * fa + pa;
        pos = fminf(fmaxf(pos, 0.0f), (float)(LL - 1));
        float fl = floorf(pos);
        int lo = (int)fl;
        int hi2 = lo + 1; if (hi2 > LL - 1) hi2 = LL - 1;
        float frac = pos - fl;
        float glo = xb[(size_t)lo * CC];
        float ghi = xb[(size_t)hi2 * CC];
        acc += wp[s] * (glo + frac * (ghi - glo));
    }
    outh[(size_t)bl * CC + h * DD + d] = f2bf(acc);
}

// ---------------------------------------------------------------------------
extern "C" void kernel_launch(void* const* d_in, const int* in_sizes, int n_in,
                              void* d_out, int out_size, void* d_ws, size_t ws_size,
                              hipStream_t stream) {
    const float* x        = (const float*)d_in[0];   // B,L,C
    const float* wave_w   = (const float*)d_in[1];   // 16 x C
    const float* wave_b   = (const float*)d_in[2];   // 16
    const float* kernel_w = (const float*)d_in[3];   // 512 x C
    const float* kernel_b = (const float*)d_in[4];   // 512
    const float* out_w    = (const float*)d_in[5];   // C x C
    float* out = (float*)d_out;

    // Workspace carve-up (256B aligned)
    char* ws = (char*)d_ws;
    size_t off = 0;
    auto carve = [&](size_t bytes) -> void* {
        void* p = (void*)(ws + off);
        off += (bytes + 255) & ~(size_t)255;
        return p;
    };
    unsigned short* xb    = (unsigned short*)carve((size_t)MM * CC * 2);   // x bf16
    unsigned short* wwb   = (unsigned short*)carve((size_t)16 * CC * 2);   // wave_w bf16
    unsigned short* kwb   = (unsigned short*)carve((size_t)512 * CC * 2);  // kernel_w bf16
    unsigned short* owb   = (unsigned short*)carve((size_t)CC * CC * 2);   // out_w bf16
    float*          wavef = (float*)carve((size_t)MM * 16 * 4);            // silu(wave)
    float*          kernf = (float*)carve((size_t)MM * 512 * 4);           // silu(kernel)
    float*          freqb = (float*)carve((size_t)MM * 4);
    float*          phaseb= (float*)carve((size_t)MM * 4);
    unsigned short* outhb = (unsigned short*)carve((size_t)MM * CC * 2);   // out_h bf16
    if (off > ws_size) return;  // workspace too small: do nothing (deterministic)

    // 1) fp32 -> bf16 conversions
    cvt_f32_to_bf16<<<4096, 256, 0, stream>>>(x,        xb,  MM * CC);
    cvt_f32_to_bf16<<<64,   256, 0, stream>>>(wave_w,   wwb, 16 * CC);
    cvt_f32_to_bf16<<<2048, 256, 0, stream>>>(kernel_w, kwb, 512 * CC);
    cvt_f32_to_bf16<<<4096, 256, 0, stream>>>(out_w,    owb, CC * CC);

    dim3 blk(256);
    // 2) wave projection: (4096 x 1024) @ (1024 x 16) -> silu
    gemm_bf16_silu<<<dim3(1, MM / 64), blk, 0, stream>>>(
        xb, wwb, wave_b, wavef, MM, 16, CC);
    // 3) kernel projection: (4096 x 1024) @ (1024 x 512) -> silu
    gemm_bf16_silu<<<dim3(2, MM / 64), blk, 0, stream>>>(
        xb, kwb, kernel_b, kernf, MM, 512, CC);
    // 4) freq/phase head means
    freq_phase_kernel<<<(MM + 255) / 256, 256, 0, stream>>>(
        wavef, freqb, phaseb, MM);
    // 5) bilinear gather + S-tap conv -> out_h (bf16)
    gather_conv_kernel<<<MM * HH, 128, 0, stream>>>(
        x, kernf, freqb, phaseb, outhb);
    // 6) output projection: (4096 x 1024) @ (1024 x 1024) -> silu -> d_out
    gemm_bf16_silu<<<dim3(4, MM / 64), blk, 0, stream>>>(
        outhb, owb, nullptr, out, MM, CC, CC);
}
